// BaseMultiHeadAttention_27522150433203
// MI455X (gfx1250) — matrix operand order repaired and checked by reading, not compiled, not run on hardware
//
#include <hip/hip_runtime.h>
#include <hip/hip_bf16.h>

// ---------------------------------------------------------------------------
// MI455X (gfx1250) fused MHA, round 2:
//  - weights pre-transposed + converted to bf16 once -> all WMMA B-fragments
//    are contiguous 32B loads (2x global_load_b128) instead of strided b32s
//  - V projection stored transposed (VT[dh][row]) so PV B-fragments are
//    contiguous as well
//  - attention K tiles staged block-wide through LDS with
//    global_load_async_to_lds_b128 + s_wait_asynccnt (CDNA5 async path)
//  - P repack C-layout -> A-layout through wave-private LDS (s_wait_dscnt)
// ---------------------------------------------------------------------------

typedef __attribute__((ext_vector_type(16))) __bf16 v16bf;
typedef __attribute__((ext_vector_type(8)))  float  v8f;

#define NH   8      // heads
#define DHD  64     // head dim
#define NQ   2048   // seq len (q == kv)
#define DM   512    // model dim == NH*DHD
#define NB   4      // batch
#define MTOT (NB * NQ)   // 8192 rows

// ---------------------------------------------------------------------------
// One-shot weight convert+transpose: WT[n][k] (bf16) = W[k][n] (f32)
// ---------------------------------------------------------------------------
__global__ __launch_bounds__(256)
void cvt_transpose_w(const float* __restrict__ W, __bf16* __restrict__ WT,
                     int K, int N) {
  const int idx = blockIdx.x * 256 + threadIdx.x;
  if (idx >= K * N) return;
  const int k = idx / N;
  const int n = idx - k * N;          // coalesced read along n
  WT[n * K + k] = (__bf16)W[idx];
}

// ---------------------------------------------------------------------------
// Projection GEMM: C = A[MxK](f32) * BT[NxK](bf16, pre-transposed).
// One wave -> 16x64 strip (4 accumulators share one A fragment).
// tout: 0 -> C[row][col] (row-major), 1 -> C[col][row] (transposed, for V).
// ---------------------------------------------------------------------------
__global__ __launch_bounds__(256)
void proj_gemm_bf16(const float* __restrict__ A, const __bf16* __restrict__ BT,
                    __bf16* __restrict__ C, int M, int K, int N, int tout) {
  const int lane = threadIdx.x & 31;
  const int wv   = threadIdx.x >> 5;
  const int gw   = blockIdx.x * 8 + wv;
  const int ntil = N / 64;
  const int tm   = (gw / ntil) * 16;
  const int tn   = (gw % ntil) * 64;
  if (tm >= M) return;
  const int lo = lane & 15;
  const int hi = lane >> 4;
  const int kb = hi * 16;       // A-frag K base (contiguous k = 16h + s)
  const int kB = hi * 16;       // B-frag K base (contiguous k = 16h + s, same as A)

  v8f acc[4];
#pragma unroll
  for (int u = 0; u < 4; ++u)
#pragma unroll
    for (int j = 0; j < 8; ++j) acc[u][j] = 0.f;

  for (int k0 = 0; k0 < K; k0 += 32) {
    v16bf a;
    const float* ap = A + (tm + lo) * K + k0 + kb;   // contiguous 16 f32
#pragma unroll
    for (int i = 0; i < 8; ++i) { a[i] = (__bf16)ap[i]; a[8 + i] = (__bf16)ap[8 + i]; }
#pragma unroll
    for (int u = 0; u < 4; ++u) {
      const __bf16* bp = BT + (size_t)(tn + u * 16 + lo) * K + k0 + kB;
      v16bf b;
#pragma unroll
      for (int i = 0; i < 16; ++i) b[i] = bp[i];     // contiguous 32B
      acc[u] = __builtin_amdgcn_wmma_f32_16x16x32_bf16(false, a, false, b,
                                                       (short)0, acc[u], false, false);
    }
  }
  if (tout == 0) {
#pragma unroll
    for (int u = 0; u < 4; ++u)
#pragma unroll
      for (int j = 0; j < 8; ++j)
        C[(size_t)(tm + j + 8 * hi) * N + tn + u * 16 + lo] = (__bf16)acc[u][j];
  } else {  // transposed store: C[col][row], contiguous along j
#pragma unroll
    for (int u = 0; u < 4; ++u)
#pragma unroll
      for (int j = 0; j < 8; ++j)
        C[(size_t)(tn + u * 16 + lo) * M + tm + j + 8 * hi] = (__bf16)acc[u][j];
  }
}

// ---------------------------------------------------------------------------
// Output GEMM: out[MxN](f32) = A[MxK](bf16) * BT[NxK](bf16) + bias[N]
// ---------------------------------------------------------------------------
__global__ __launch_bounds__(256)
void out_gemm_f32(const __bf16* __restrict__ A, const __bf16* __restrict__ BT,
                  const float* __restrict__ bias, float* __restrict__ C,
                  int M, int K, int N) {
  const int lane = threadIdx.x & 31;
  const int wv   = threadIdx.x >> 5;
  const int gw   = blockIdx.x * 8 + wv;
  const int ntil = N / 64;
  const int tm   = (gw / ntil) * 16;
  const int tn   = (gw % ntil) * 64;
  if (tm >= M) return;
  const int lo = lane & 15, hi = lane >> 4;
  const int kb = hi * 16, kB = hi * 16;

  v8f acc[4];
#pragma unroll
  for (int u = 0; u < 4; ++u)
#pragma unroll
    for (int j = 0; j < 8; ++j) acc[u][j] = 0.f;

  for (int k0 = 0; k0 < K; k0 += 32) {
    v16bf a;
    const __bf16* ap = A + (size_t)(tm + lo) * K + k0 + kb;
#pragma unroll
    for (int i = 0; i < 8; ++i) { a[i] = ap[i]; a[8 + i] = ap[8 + i]; }
#pragma unroll
    for (int u = 0; u < 4; ++u) {
      const __bf16* bp = BT + (size_t)(tn + u * 16 + lo) * K + k0 + kB;
      v16bf b;
#pragma unroll
      for (int i = 0; i < 16; ++i) b[i] = bp[i];
      acc[u] = __builtin_amdgcn_wmma_f32_16x16x32_bf16(false, a, false, b,
                                                       (short)0, acc[u], false, false);
    }
  }
#pragma unroll
  for (int u = 0; u < 4; ++u)
#pragma unroll
    for (int j = 0; j < 8; ++j) {
      const int c = tn + u * 16 + lo;
      C[(size_t)(tm + j + 8 * hi) * N + c] = acc[u][j] + bias[c];
    }
}

// ---------------------------------------------------------------------------
// Flash attention with diagonal logit. One wave per (b, h, 16-q tile);
// all 8 waves of a block share (b,h) -> K tile staged block-wide in LDS
// via async-to-LDS loads. V read from transposed VT[dh][row] (contiguous).
// ---------------------------------------------------------------------------
__global__ __launch_bounds__(256)
void flash_attn(const __bf16* __restrict__ Qb, const __bf16* __restrict__ Kb,
                const __bf16* __restrict__ VT, const __bf16* __restrict__ QKb,
                const __bf16* __restrict__ QVb, __bf16* __restrict__ Ob) {
  __shared__ __bf16 Ktile[32 * 64];      // 32 keys x 64 dh (4 KiB), block-shared
  __shared__ __bf16 ptile[8][16 * 32];   // per-wave P staging (8 KiB)

  const int lane = threadIdx.x & 31;
  const int wv   = threadIdx.x >> 5;
  const int gw   = blockIdx.x * 8 + wv;        // 0..4095
  const int qt   = gw & 127;                   // 128 q-tiles of 16
  const int bh   = gw >> 7;                    // same for all 8 waves in block
  const int bidx = bh >> 3;
  const int h    = bh & 7;
  const int lo = lane & 15, hi = lane >> 4;
  const int kb = hi * 16, kB = hi * 16;
  const float scale = 0.125f;                  // DH^-0.5

  const __bf16* Qh = Qb + (size_t)(bidx * NQ) * DM + h * DHD;
  const __bf16* Kh = Kb + (size_t)(bidx * NQ) * DM + h * DHD;

  // per-thread K-tile staging slot: key = tid>>3, 16B segment = tid&7
  const int skey = threadIdx.x >> 3;
  const int sseg = threadIdx.x & 7;
  const unsigned klds = (unsigned)(size_t)&Ktile[skey * 64 + sseg * 8];

  // Q A-fragments (dh 0..31, 32..63) for rows qt*16..+15
  v16bf qa0, qa1;
  {
    const __bf16* qp = Qh + (size_t)(qt * 16 + lo) * DM;
#pragma unroll
    for (int i = 0; i < 8; ++i) { qa0[i] = qp[kb + i];      qa0[8 + i] = qp[kb + 8 + i]; }
#pragma unroll
    for (int i = 0; i < 8; ++i) { qa1[i] = qp[32 + kb + i]; qa1[8 + i] = qp[32 + kb + 8 + i]; }
  }

  v8f o[4];
  float mrow[8], lrow[8];
#pragma unroll
  for (int u = 0; u < 4; ++u)
#pragma unroll
    for (int j = 0; j < 8; ++j) o[u][j] = 0.f;
#pragma unroll
  for (int j = 0; j < 8; ++j) { mrow[j] = -1e30f; lrow[j] = 0.f; }

  for (int key0 = 0; key0 < NQ; key0 += 32) {
    // ---- async stage K tile (32x64 bf16 = 4KB): one b128 per thread ----
    {
      const unsigned long long kga =
          (unsigned long long)(size_t)(Kh + (size_t)(key0 + skey) * DM + sseg * 8);
      asm volatile("global_load_async_to_lds_b128 %0, %1, off"
                   :: "v"(klds), "v"(kga) : "memory");
      asm volatile("s_wait_asynccnt 0" ::: "memory");
    }
    __syncthreads();

    // ---- S = scale * Q Kt : two 16x16 tiles from LDS K tile ----
    v8f s[2];
#pragma unroll
    for (int t = 0; t < 2; ++t) {
      v8f c;
#pragma unroll
      for (int j = 0; j < 8; ++j) c[j] = 0.f;
      const __bf16* kp = &Ktile[(t * 16 + lo) * 64];   // key row, contiguous dh
      v16bf b0, b1;
#pragma unroll
      for (int i = 0; i < 16; ++i) b0[i] = kp[kB + i];        // dh 0..31
#pragma unroll
      for (int i = 0; i < 16; ++i) b1[i] = kp[32 + kB + i];   // dh 32..63
      c = __builtin_amdgcn_wmma_f32_16x16x32_bf16(false, qa0, false, b0, (short)0, c, false, false);
      c = __builtin_amdgcn_wmma_f32_16x16x32_bf16(false, qa1, false, b1, (short)0, c, false, false);
      s[t] = c;
    }

    // ---- online softmax over the 32 new keys ----
#pragma unroll
    for (int j = 0; j < 8; ++j) {
      float v0 = s[0][j] * scale, v1 = s[1][j] * scale;
      float mx = fmaxf(v0, v1);
      mx = fmaxf(mx, __shfl_xor(mx, 1, 32));
      mx = fmaxf(mx, __shfl_xor(mx, 2, 32));
      mx = fmaxf(mx, __shfl_xor(mx, 4, 32));
      mx = fmaxf(mx, __shfl_xor(mx, 8, 32));
      const float nm   = fmaxf(mrow[j], mx);
      const float corr = __expf(mrow[j] - nm);
      mrow[j] = nm;
      const float p0 = __expf(v0 - nm), p1 = __expf(v1 - nm);
      float ps = p0 + p1;
      ps += __shfl_xor(ps, 1, 32);
      ps += __shfl_xor(ps, 2, 32);
      ps += __shfl_xor(ps, 4, 32);
      ps += __shfl_xor(ps, 8, 32);
      lrow[j] = lrow[j] * corr + ps;
#pragma unroll
      for (int u = 0; u < 4; ++u) o[u][j] *= corr;
      const int m = j + 8 * hi;                       // C-layout row
      ptile[wv][m * 32 + lo]      = (__bf16)p0;
      ptile[wv][m * 32 + 16 + lo] = (__bf16)p1;
    }
    // same-wave LDS RAW ordering for the P repack
    asm volatile("s_wait_dscnt 0" ::: "memory");

    // ---- O += P (16x32) * V (32x64), V fragments contiguous from VT ----
    v16bf pa;
    {
      const __bf16* pp = &ptile[wv][lo * 32 + kb];    // A-layout row lo
#pragma unroll
      for (int i = 0; i < 8; ++i) { pa[i] = pp[i]; pa[8 + i] = pp[8 + i]; }
    }
#pragma unroll
    for (int u = 0; u < 4; ++u) {
      const __bf16* vp = VT + (size_t)(h * DHD + u * 16 + lo) * MTOT
                            + bidx * NQ + key0 + kB;
      v16bf vb;
#pragma unroll
      for (int i = 0; i < 16; ++i) vb[i] = vp[i];     // contiguous 32B
      o[u] = __builtin_amdgcn_wmma_f32_16x16x32_bf16(false, pa, false, vb,
                                                     (short)0, o[u], false, false);
    }
    __syncthreads();   // protect K tile before next iteration's async store
  }

  // ---- diagonal term: extra logit scale*dot(q, xq_k), value xq_v ----
  {
    const int row = qt * 16 + lo;                     // both halves cover rows 0..15
    const __bf16* qp = Qh + (size_t)row * DM;
    const __bf16* dk = QKb + (size_t)(bidx * NQ + row) * DM + h * DHD;
    float dot = 0.f;
#pragma unroll
    for (int i = 0; i < DHD; ++i) dot += (float)qp[i] * (float)dk[i];
    const float dlog = dot * scale;
#pragma unroll
    for (int j = 0; j < 8; ++j) {
      const float dj   = __shfl(dlog, j + 8 * hi, 32);
      const float nm   = fmaxf(mrow[j], dj);
      const float corr = __expf(mrow[j] - nm);
      const float pd   = __expf(dj - nm);
      mrow[j] = nm;
      lrow[j] = lrow[j] * corr + pd;
      const int r = qt * 16 + j + 8 * hi;
#pragma unroll
      for (int u = 0; u < 4; ++u) {
        const float xv = (float)QVb[(size_t)(bidx * NQ + r) * DM + h * DHD + u * 16 + lo];
        o[u][j] = o[u][j] * corr + pd * xv;
      }
    }
  }

  // ---- normalize and store (merged-head layout [b, q, h*DHD+dh]) ----
#pragma unroll
  for (int j = 0; j < 8; ++j) {
    const float inv = 1.f / lrow[j];
    const int r = qt * 16 + j + 8 * hi;
#pragma unroll
    for (int u = 0; u < 4; ++u)
      Ob[(size_t)(bidx * NQ + r) * DM + h * DHD + u * 16 + lo] = (__bf16)(o[u][j] * inv);
  }
}

// ---------------------------------------------------------------------------
extern "C" void kernel_launch(void* const* d_in, const int* in_sizes, int n_in,
                              void* d_out, int out_size, void* d_ws, size_t ws_size,
                              hipStream_t stream) {
  (void)in_sizes; (void)n_in; (void)out_size; (void)ws_size;
  const float* xq = (const float*)d_in[0];
  const float* xk = (const float*)d_in[1];
  const float* xv = (const float*)d_in[2];
  const float* Wq = (const float*)d_in[3];
  const float* Wk = (const float*)d_in[4];
  const float* Wv = (const float*)d_in[5];
  const float* Wo = (const float*)d_in[6];
  const float* bo = (const float*)d_in[7];
  float* out = (float*)d_out;

  const size_t BUF  = (size_t)MTOT * DM * 2;   // 8 MiB bf16 activation buffer
  const size_t WBUF = (size_t)DM * DM * 2;     // 512 KiB bf16 weight buffer
  char* ws = (char*)d_ws;
  __bf16* Qb   = (__bf16*)(ws + 0 * BUF);
  __bf16* Kb   = (__bf16*)(ws + 1 * BUF);
  __bf16* Vt   = (__bf16*)(ws + 2 * BUF);      // transposed: [DM][MTOT]
  __bf16* QKb  = (__bf16*)(ws + 3 * BUF);
  __bf16* QVb  = (__bf16*)(ws + 4 * BUF);
  __bf16* Ab   = (__bf16*)(ws + 5 * BUF);
  __bf16* WqT  = (__bf16*)(ws + 6 * BUF + 0 * WBUF);
  __bf16* WkT  = (__bf16*)(ws + 6 * BUF + 1 * WBUF);
  __bf16* WvT  = (__bf16*)(ws + 6 * BUF + 2 * WBUF);
  __bf16* WoT  = (__bf16*)(ws + 6 * BUF + 3 * WBUF);

  dim3 blk(256);
  dim3 gcvt((DM * DM) / 256);                  // 1024 blocks
  hipLaunchKernelGGL(cvt_transpose_w, gcvt, blk, 0, stream, Wq, WqT, DM, DM);
  hipLaunchKernelGGL(cvt_transpose_w, gcvt, blk, 0, stream, Wk, WkT, DM, DM);
  hipLaunchKernelGGL(cvt_transpose_w, gcvt, blk, 0, stream, Wv, WvT, DM, DM);
  hipLaunchKernelGGL(cvt_transpose_w, gcvt, blk, 0, stream, Wo, WoT, DM, DM);

  dim3 gproj((MTOT / 16) * (DM / 64) / 8);     // 512 blocks, 8 waves each
  hipLaunchKernelGGL(proj_gemm_bf16, gproj, blk, 0, stream, xq, WqT, Qb,  MTOT, DM, DM, 0);
  hipLaunchKernelGGL(proj_gemm_bf16, gproj, blk, 0, stream, xk, WkT, Kb,  MTOT, DM, DM, 0);
  hipLaunchKernelGGL(proj_gemm_bf16, gproj, blk, 0, stream, xv, WvT, Vt,  MTOT, DM, DM, 1);
  hipLaunchKernelGGL(proj_gemm_bf16, gproj, blk, 0, stream, xq, WkT, QKb, MTOT, DM, DM, 0);
  hipLaunchKernelGGL(proj_gemm_bf16, gproj, blk, 0, stream, xq, WvT, QVb, MTOT, DM, DM, 0);

  dim3 gattn((NB * NH * (NQ / 16)) / 8);       // 4096 waves -> 512 blocks
  hipLaunchKernelGGL(flash_attn, gattn, blk, 0, stream, Qb, Kb, Vt, QKb, QVb, Ab);

  hipLaunchKernelGGL(out_gemm_f32, gproj, blk, 0, stream, Ab, WoT, bo, out, MTOT, DM, DM);
}
